// meta_model_72842645340795
// MI455X (gfx1250) — matrix operand order repaired and checked by reading, not compiled, not run on hardware
//
#include <hip/hip_runtime.h>

#define N_NODES 8192
#define FEAT    512
#define HID     128

typedef __attribute__((ext_vector_type(16))) _Float16 v16h;
typedef __attribute__((ext_vector_type(8)))  float    v8f;

#ifndef __has_builtin
#define __has_builtin(x) 0
#endif

#if __has_builtin(__builtin_amdgcn_tensor_load_to_lds)
#define HAVE_TDM 1
#else
#define HAVE_TDM 0
#endif

static __device__ __forceinline__ v8f zero8() { v8f z = {}; return z; }

static __device__ __forceinline__ v8f wmma_f16(v16h a, v16h b, v8f c) {
  return __builtin_amdgcn_wmma_f32_16x16x32_f16(false, a, false, b, (short)0, c,
                                                false, false);
}

// A-matrix 16x32 f16 fragment from row-major f16 (ld = row stride).
// Lane (r, hi): halves [0..7] = A[r][kb+8*hi ..], [8..15] = A[r][kb+16+8*hi ..]
static __device__ __forceinline__ v16h frag_A_f16(const _Float16* base, int ld,
                                                  int r, int hi, int kb) {
  const _Float16* p = base + (size_t)r * ld + kb + hi * 8;
  v16h a;
#pragma unroll
  for (int i = 0; i < 8; ++i) a[i] = p[i];
#pragma unroll
  for (int i = 0; i < 8; ++i) a[8 + i] = p[16 + i];
  return a;
}

// Same fragment sourced from f32 memory (cvt to f16 on load).
static __device__ __forceinline__ v16h frag_A_f32(const float* base, int ld,
                                                  int r, int hi, int kb) {
  const float* p = base + (size_t)r * ld + kb + hi * 8;
  v16h a;
#pragma unroll
  for (int i = 0; i < 8; ++i) a[i] = (_Float16)p[i];
#pragma unroll
  for (int i = 0; i < 8; ++i) a[8 + i] = (_Float16)p[16 + i];
  return a;
}

// B-matrix 32x16 fragment where B[k][n] = M[n][k], M row-major f16 (ld cols).
// Lane (n, hi): halves [0..7] = M[n][kb+8*hi ..], [8..15] = M[n][kb+16+8*hi ..]
static __device__ __forceinline__ v16h frag_B_t(const _Float16* M, int ld,
                                                int n_row, int hi, int kb) {
  const _Float16* p = M + (size_t)n_row * ld + kb + hi * 8;
  v16h b;
#pragma unroll
  for (int i = 0; i < 8; ++i) b[i] = p[i];
#pragma unroll
  for (int i = 0; i < 8; ++i) b[8 + i] = p[16 + i];
  return b;
}

static __device__ __forceinline__ float sigmoidf(float x) {
  return 1.0f / (1.0f + __expf(-x));
}

template <int N>
static __device__ __forceinline__ void wait_tensor() {
#if __has_builtin(__builtin_amdgcn_s_wait_tensorcnt)
  __builtin_amdgcn_s_wait_tensorcnt((short)N);
#endif
}

#if HAVE_TDM
typedef __attribute__((ext_vector_type(4))) unsigned int u32x4;
typedef __attribute__((ext_vector_type(8))) int i32x8;
typedef __attribute__((ext_vector_type(4))) int i32x4;

// 2D f16 tile DMA: global (row-major, row_stride elems) -> LDS, via the
// Tensor Data Mover. Tracked on TENSORcnt.
static __device__ __forceinline__ void tdm_load_tile_f16(
    unsigned lds_byte_off, const void* gsrc, int tile_rows, int tile_cols,
    int tensor_rows, int row_stride) {
  unsigned long long ga = (unsigned long long)(uintptr_t)gsrc;
  u32x4 g0;
  g0[0] = 1u;                                             // count=1 (valid D#)
  g0[1] = lds_byte_off;                                   // lds_addr
  g0[2] = (unsigned)ga;                                   // global_addr[31:0]
  g0[3] = (unsigned)((ga >> 32) & 0x01ffffffu) | (2u << 30);  // addr[56:32] | type=2
  i32x8 g1;
  g1[0] = (int)(1u << 16);                                // data_size = 2 bytes
  g1[1] = (int)(((unsigned)tile_cols & 0xffffu) << 16);   // tensor_dim0 lo16
  g1[2] = (int)(((unsigned)tensor_rows & 0xffffu) << 16); // dim0 hi=0 | dim1 lo16
  g1[3] = (int)((((unsigned)tensor_rows >> 16) & 0xffffu) |
                (((unsigned)tile_cols & 0xffffu) << 16)); // dim1 hi | tile_dim0
  g1[4] = (int)((unsigned)tile_rows & 0xffffu);           // tile_dim1 (tile_dim2=0)
  g1[5] = row_stride;                                     // tensor_dim0_stride lo32
  g1[6] = 0;
  g1[7] = 0;
  i32x4 z4 = {};
#if __clang_major__ >= 23
  i32x8 z8 = {};
  __builtin_amdgcn_tensor_load_to_lds(g0, g1, z4, z4, z8, 0);
#else
  __builtin_amdgcn_tensor_load_to_lds(g0, g1, z4, z4, 0);
#endif
}
#endif  // HAVE_TDM

// ---------------------------------------------------------------------------
// Kernel 0: pack weights to f16, transposed (W1t[m][n][k], W2t[m][n][k]),
// biases copied to contiguous [4][128]. MLP order: link, aa, mod, unlink.
// ---------------------------------------------------------------------------
__global__ __launch_bounds__(256) void pack_kernel(
    const float* __restrict__ Wl1, const float* __restrict__ Wa1,
    const float* __restrict__ Wm1, const float* __restrict__ Wu1,
    const float* __restrict__ Wl2, const float* __restrict__ Wa2,
    const float* __restrict__ Wm2, const float* __restrict__ Wu2,
    const float* __restrict__ bl1, const float* __restrict__ ba1,
    const float* __restrict__ bm1, const float* __restrict__ bu1,
    const float* __restrict__ bl2, const float* __restrict__ ba2,
    const float* __restrict__ bm2, const float* __restrict__ bu2,
    _Float16* __restrict__ W1t, _Float16* __restrict__ W2t,
    float* __restrict__ b1o, float* __restrict__ b2o) {
  const float* W1s[4] = {Wl1, Wa1, Wm1, Wu1};
  const float* W2s[4] = {Wl2, Wa2, Wm2, Wu2};
  const float* b1s[4] = {bl1, ba1, bm1, bu1};
  const float* b2s[4] = {bl2, ba2, bm2, bu2};
  const int tid = blockIdx.x * 256 + threadIdx.x;
  const int n1 = 4 * HID * FEAT;   // 262144
  const int n2 = 4 * HID * HID;    // 65536
  if (tid < n1) {
    int m = tid >> 16, rem = tid & 0xffff;
    int n = rem >> 9, k = rem & 511;           // [n][k] out
    W1t[tid] = (_Float16)W1s[m][(size_t)k * HID + n];
  } else if (tid < n1 + n2) {
    int t = tid - n1;
    int m = t >> 14, rem = t & 0x3fff;
    int n = rem >> 7, k = rem & 127;
    W2t[t] = (_Float16)W2s[m][(size_t)k * HID + n];
  } else if (tid < n1 + n2 + 4 * HID) {
    int t = tid - (n1 + n2);
    b1o[t] = b1s[t >> 7][t & 127];
  } else if (tid < n1 + n2 + 8 * HID) {
    int t = tid - (n1 + n2 + 4 * HID);
    b2o[t] = b2s[t >> 7][t & 127];
  }
}

// ---------------------------------------------------------------------------
// Kernel 1: four 2-layer MLPs -> f16 latents [4][N][HID]
// grid = N/16 blocks of 128 threads; wave w handles MLP w for one 16-row tile.
// ---------------------------------------------------------------------------
__global__ __launch_bounds__(128) void latent_kernel(
    const float* __restrict__ feat,
    const _Float16* __restrict__ W1t, const _Float16* __restrict__ W2t,
    const float* __restrict__ b1p, const float* __restrict__ b2p,
    _Float16* __restrict__ lat_out) {
  __shared__ _Float16 hbuf[4][16][HID];
  const int w = threadIdx.x >> 5;
  const int lane = threadIdx.x & 31;
  const int hi = lane >> 4, r = lane & 15;

  const _Float16* W1 = W1t + (size_t)w * HID * FEAT;  // [128][512]
  const _Float16* W2 = W2t + (size_t)w * HID * HID;   // [128][128]
  const float* B1 = b1p + w * HID;
  const float* B2 = b2p + w * HID;

  const int i0 = blockIdx.x * 16;
  const float* featT = feat + (size_t)i0 * FEAT;

  // ---- h = relu(feat @ W1 + b1) : 16x128, K=512 ----
  v8f c1[8];
#pragma unroll
  for (int nt = 0; nt < 8; ++nt) c1[nt] = zero8();
#pragma unroll 4
  for (int kk = 0; kk < FEAT / 32; ++kk) {
    const int kb = kk * 32;
    v16h a = frag_A_f32(featT, FEAT, r, hi, kb);
#pragma unroll
    for (int nt = 0; nt < 8; ++nt) {
      v16h b = frag_B_t(W1, FEAT, nt * 16 + r, hi, kb);
      c1[nt] = wmma_f16(a, b, c1[nt]);
    }
  }
#pragma unroll
  for (int nt = 0; nt < 8; ++nt) {
    const int col = nt * 16 + r;
    const float bias = B1[col];
#pragma unroll
    for (int v = 0; v < 8; ++v) {
      float x = c1[nt][v] + bias;
      hbuf[w][v + 8 * hi][col] = (_Float16)(x > 0.0f ? x : 0.0f);
    }
  }
  __syncthreads();

  // ---- lat = h @ W2 + b2 : 16x128, K=128 ----
  v8f c2[8];
#pragma unroll
  for (int nt = 0; nt < 8; ++nt) c2[nt] = zero8();
#pragma unroll
  for (int kk = 0; kk < HID / 32; ++kk) {
    const int kb = kk * 32;
    v16h a = frag_A_f16(&hbuf[w][0][0], HID, r, hi, kb);
#pragma unroll
    for (int nt = 0; nt < 8; ++nt) {
      v16h b = frag_B_t(W2, HID, nt * 16 + r, hi, kb);
      c2[nt] = wmma_f16(a, b, c2[nt]);
    }
  }
  _Float16* latg = lat_out + (size_t)w * N_NODES * HID;
#pragma unroll
  for (int nt = 0; nt < 8; ++nt) {
    const int col = nt * 16 + r;
    const float bias = B2[col];
#pragma unroll
    for (int v = 0; v < 8; ++v) {
      latg[(size_t)(i0 + v + 8 * hi) * HID + col] = (_Float16)(c2[nt][v] + bias);
    }
  }
}

// ---------------------------------------------------------------------------
// Kernel 2: fused gram(x4) + sigmoid + weighted combine + adj blend.
// grid (64,64), 256 threads (8 waves), 128x128 block tile, 32x64 wave tile.
// j-side latent rows (128 x HID f16 per gram) are TDM-DMA'd into LDS,
// double-buffered across the 4 grams. A-side fragments stream from L2.
// ---------------------------------------------------------------------------
static __device__ __forceinline__ void gram_g(const _Float16* __restrict__ Ag,
                                              const _Float16* Bl, int i0,
                                              int jloc, int r, int hi,
                                              v8f S[2][4]) {
#pragma unroll
  for (int mt = 0; mt < 2; ++mt)
#pragma unroll
    for (int nt = 0; nt < 4; ++nt) S[mt][nt] = zero8();
#pragma unroll
  for (int kk = 0; kk < HID / 32; ++kk) {
    const int kb = kk * 32;
    v16h a[2];
#pragma unroll
    for (int mt = 0; mt < 2; ++mt)
      a[mt] = frag_A_f16(Ag + (size_t)(i0 + mt * 16) * HID, HID, r, hi, kb);
    v16h b[4];
#pragma unroll
    for (int nt = 0; nt < 4; ++nt)
      b[nt] = frag_B_t(Bl, HID, jloc + nt * 16 + r, hi, kb);
#pragma unroll
    for (int mt = 0; mt < 2; ++mt)
#pragma unroll
      for (int nt = 0; nt < 4; ++nt) S[mt][nt] = wmma_f16(a[mt], b[nt], S[mt][nt]);
  }
}

__global__ __launch_bounds__(256) void ctx_kernel(
    const _Float16* __restrict__ lat,  // [4][N][HID]: link, aa, mod, unlink
    const float* __restrict__ adj,
    const float* __restrict__ aa,
    const float* __restrict__ mod,
    const float* __restrict__ wsum,
    float* __restrict__ out) {
  __shared__ _Float16 Bbuf[2][128][HID];  // 64 KB, double-buffered j-side tile

  const int w    = threadIdx.x >> 5;
  const int lane = threadIdx.x & 31;
  const int wm   = w >> 1;
  const int wn   = w & 1;
  const int hi   = lane >> 4;
  const int r    = lane & 15;

  const int i0   = blockIdx.y * 128 + wm * 32;  // global row base of wave tile
  const int jblk = blockIdx.x * 128;            // block's column base
  const int jloc = wn * 64;                     // wave's column base in LDS tile

  // Prefetch streamed tiles into L2 (global_prefetch_b8).
  {
    size_t pidx = (size_t)(i0 + lane) * N_NODES + jblk;
    __builtin_prefetch(aa + pidx, 0, 0);
    __builtin_prefetch(mod + pidx, 0, 0);
    __builtin_prefetch(adj + pidx, 0, 0);
  }

  // Stage gram g's j-rows into Bbuf[buf]; each wave DMAs its own 16 rows.
  auto issue_stage = [&](int g, int buf) {
    const _Float16* src =
        lat + ((size_t)g * N_NODES + (size_t)jblk + (size_t)w * 16) * HID;
#if HAVE_TDM
    unsigned ldsoff = (unsigned)(uintptr_t)(const void*)&Bbuf[buf][w * 16][0];
    tdm_load_tile_f16(ldsoff, src, 16, HID, N_NODES, HID);
#else
    uint4* d = reinterpret_cast<uint4*>(&Bbuf[buf][w * 16][0]);
    const uint4* s = reinterpret_cast<const uint4*>(src);
#pragma unroll
    for (int it = 0; it < 8; ++it) d[it * 32 + lane] = s[it * 32 + lane];
#endif
  };

  // softmax(ws[0..2])
  const float e0 = __expf(wsum[0]);
  const float e1 = __expf(wsum[1]);
  const float e2 = __expf(wsum[2]);
  const float inv = 1.0f / (e0 + e1 + e2);
  const float w0 = e0 * inv, w1 = e1 * inv, w2 = e2 * inv;

  v8f F[2][4];
  v8f S[2][4];

  issue_stage(0, 0);
  issue_stage(1, 1);

  // ---- g0: link ----
  wait_tensor<1>();
  __syncthreads();
  gram_g(lat, &Bbuf[0][0][0], i0, jloc, r, hi, S);
#pragma unroll
  for (int mt = 0; mt < 2; ++mt)
#pragma unroll
    for (int nt = 0; nt < 4; ++nt)
#pragma unroll
      for (int v = 0; v < 8; ++v) F[mt][nt][v] = w0 * sigmoidf(S[mt][nt][v]);
  __syncthreads();
  issue_stage(2, 0);

  // ---- g1: aa ----
  wait_tensor<1>();
  __syncthreads();
  gram_g(lat + (size_t)1 * N_NODES * HID, &Bbuf[1][0][0], i0, jloc, r, hi, S);
#pragma unroll
  for (int mt = 0; mt < 2; ++mt)
#pragma unroll
    for (int nt = 0; nt < 4; ++nt) {
      const int col = jblk + jloc + nt * 16 + r;
#pragma unroll
      for (int v = 0; v < 8; ++v) {
        const size_t idx = (size_t)(i0 + mt * 16 + v + 8 * hi) * N_NODES + col;
        const float m = __builtin_nontemporal_load(aa + idx);
        F[mt][nt][v] += w1 * sigmoidf(S[mt][nt][v] * m);
      }
    }
  __syncthreads();
  issue_stage(3, 1);

  // ---- g2: mod ----
  wait_tensor<1>();
  __syncthreads();
  gram_g(lat + (size_t)2 * N_NODES * HID, &Bbuf[0][0][0], i0, jloc, r, hi, S);
#pragma unroll
  for (int mt = 0; mt < 2; ++mt)
#pragma unroll
    for (int nt = 0; nt < 4; ++nt) {
      const int col = jblk + jloc + nt * 16 + r;
#pragma unroll
      for (int v = 0; v < 8; ++v) {
        const size_t idx = (size_t)(i0 + mt * 16 + v + 8 * hi) * N_NODES + col;
        const float m = __builtin_nontemporal_load(mod + idx);
        F[mt][nt][v] += w2 * sigmoidf(S[mt][nt][v] * m);
      }
    }

  // ---- g3: unlink + final blend ----
  wait_tensor<0>();
  __syncthreads();
  gram_g(lat + (size_t)3 * N_NODES * HID, &Bbuf[1][0][0], i0, jloc, r, hi, S);
#pragma unroll
  for (int mt = 0; mt < 2; ++mt)
#pragma unroll
    for (int nt = 0; nt < 4; ++nt) {
      const int col = jblk + jloc + nt * 16 + r;
#pragma unroll
      for (int v = 0; v < 8; ++v) {
        const size_t idx = (size_t)(i0 + mt * 16 + v + 8 * hi) * N_NODES + col;
        const float u = sigmoidf(S[mt][nt][v]);
        const float a = __builtin_nontemporal_load(adj + idx);
        __builtin_nontemporal_store((F[mt][nt][v] - u) * a + u, out + idx);
      }
    }
}

// ---------------------------------------------------------------------------
extern "C" void kernel_launch(void* const* d_in, const int* in_sizes, int n_in,
                              void* d_out, int out_size, void* d_ws, size_t ws_size,
                              hipStream_t stream) {
  const float* feat = (const float*)d_in[0];
  const float* adj  = (const float*)d_in[1];
  const float* aa   = (const float*)d_in[2];
  const float* mod  = (const float*)d_in[3];
  const float* Wl1 = (const float*)d_in[4],  *bl1 = (const float*)d_in[5];
  const float* Wl2 = (const float*)d_in[6],  *bl2 = (const float*)d_in[7];
  const float* Wu1 = (const float*)d_in[8],  *bu1 = (const float*)d_in[9];
  const float* Wu2 = (const float*)d_in[10], *bu2 = (const float*)d_in[11];
  const float* Wa1 = (const float*)d_in[12], *ba1 = (const float*)d_in[13];
  const float* Wa2 = (const float*)d_in[14], *ba2 = (const float*)d_in[15];
  const float* Wm1 = (const float*)d_in[16], *bm1 = (const float*)d_in[17];
  const float* Wm2 = (const float*)d_in[18], *bm2 = (const float*)d_in[19];
  const float* ws  = (const float*)d_in[20];

  // Workspace layout (bytes):
  char* wsb = (char*)d_ws;
  _Float16* lat = (_Float16*)(wsb);                 // 4*8192*128*2 = 8388608
  _Float16* W1t = (_Float16*)(wsb + 8388608);       // 4*128*512*2  = 524288
  _Float16* W2t = (_Float16*)(wsb + 8912896);       // 4*128*128*2  = 131072
  float*    b1p = (float*)(wsb + 9043968);          // 4*128*4      = 2048
  float*    b2p = (float*)(wsb + 9046016);          // 4*128*4      = 2048

  const int pack_elems = 4 * HID * FEAT + 4 * HID * HID + 8 * HID;
  pack_kernel<<<dim3((pack_elems + 255) / 256), dim3(256), 0, stream>>>(
      Wl1, Wa1, Wm1, Wu1, Wl2, Wa2, Wm2, Wu2,
      bl1, ba1, bm1, bu1, bl2, ba2, bm2, bu2,
      W1t, W2t, b1p, b2p);

  latent_kernel<<<dim3(N_NODES / 16), dim3(128), 0, stream>>>(
      feat, W1t, W2t, b1p, b2p, lat);

  ctx_kernel<<<dim3(N_NODES / 128, N_NODES / 128), dim3(256), 0, stream>>>(
      lat, adj, aa, mod, ws, (float*)d_out);
}
